// NeuralODE_82575041233356
// MI455X (gfx1250) — matrix operand order repaired and checked by reading, not compiled, not run on hardware
//
#include <hip/hip_runtime.h>
#include <hip/hip_bf16.h>

// Neural ODE, RK4, fully fused persistent integration on gfx1250 WMMA.
//   f(y) = W2 @ ELU(W1 @ y + b1) + b2,  y: [B=262144, 64], H=256, 64 RK4 steps.
// v4: best-of-v2/v3. All W1/W2 WMMA B-fragments live in per-lane VGPRs (512
// regs) so the 256 f-evals do zero LDS traffic for weights; biases ride in the
// WMMA C operand (splat from hoisted per-lane registers), which gave the
// cleanest epilogue codegen (paired cvt_pk_bf16 + ds_store_b16/_d16_hi). LDS
// is used only for the two inherent D->A fragment transposes per f-eval.

#define D_IN   64
#define H_DIM  256
#define BATCH  262144
#define NSTEPS 64

typedef __attribute__((ext_vector_type(16))) __bf16 bf16x16;
typedef __attribute__((ext_vector_type(8)))  __bf16 bf16x8;
typedef __attribute__((ext_vector_type(8)))  float  f32x8;

// LDS layout (bytes)
#define OFF_W1 0                         // 256x64 bf16   = 32768
#define OFF_W2 32768                     // 64x256 bf16   = 32768
#define OFF_B1 65536                     // 256 f32       = 1024
#define OFF_B2 66560                     // 64 f32        = 256
#define OFF_TS 66816                     // 4 waves * 16x64 bf16 (2048 B each)
#define OFF_HS (66816 + 4 * 2048)        // 4 waves * 16x256 bf16 (8192 B each)
#define SMEM_BYTES (OFF_HS + 4 * 8192)   // 107776 B, fits 320 KB/WGP

__global__ void __launch_bounds__(128)
node_rk4_kernel(const float* __restrict__ x, const float* __restrict__ t,
                const float* __restrict__ W1, const float* __restrict__ b1,
                const float* __restrict__ W2, const float* __restrict__ b2,
                float* __restrict__ out)
{
    extern __shared__ char smem[];
    __bf16* W1s = (__bf16*)(smem + OFF_W1);   // [H][D] row-major, bf16
    __bf16* W2s = (__bf16*)(smem + OFF_W2);   // [D][H] row-major, bf16
    float*  b1s = (float*)(smem + OFF_B1);
    float*  b2s = (float*)(smem + OFF_B2);

    const int tid  = threadIdx.x;
    const int wave = tid >> 5;
    const int lane = tid & 31;
    const int ln   = lane & 15;     // column index within 16-wide tile (C/D/B lanes)
    const int hi   = lane >> 4;     // half-wave select

    __bf16* ts = (__bf16*)(smem + OFF_TS) + wave * (16 * D_IN);   // stage input tile [16][64]
    __bf16* hs = (__bf16*)(smem + OFF_HS) + wave * (16 * H_DIM);  // hidden tile     [16][256]

    // Cooperative weight load + f32 -> bf16 conversion (once).
    for (int i = tid; i < H_DIM * D_IN; i += 128) W1s[i] = (__bf16)W1[i];
    for (int i = tid; i < D_IN * H_DIM; i += 128) W2s[i] = (__bf16)W2[i];
    for (int i = tid; i < H_DIM; i += 128)        b1s[i] = b1[i];
    if (tid < D_IN)                               b2s[tid] = b2[tid];
    __syncthreads();

    // ---- Hoist all weight B-fragments into registers (loop-invariant). ----
    // B fragment (32x16 bf16): lane = output col; K = kc*32 + hi*16 + e (contiguous 32B).
    bf16x16 w1f[16][2];   // GEMM1: 16 H-tiles x 2 K-chunks  (256 VGPRs)
#pragma unroll
    for (int nt = 0; nt < 16; ++nt)
#pragma unroll
        for (int kc = 0; kc < 2; ++kc)
            w1f[nt][kc] = *(const bf16x16*)(W1s + (nt * 16 + ln) * D_IN + kc * 32 + hi * 16);

    bf16x16 w2f[8][4];    // GEMM2: 8 K-chunks x 4 D-tiles   (256 VGPRs)
#pragma unroll
    for (int kc = 0; kc < 8; ++kc)
#pragma unroll
        for (int nt = 0; nt < 4; ++nt)
            w2f[kc][nt] = *(const bf16x16*)(W2s + (nt * 16 + ln) * H_DIM + kc * 32 + hi * 16);

    // ---- Hoist per-lane bias scalars (loop-invariant). ----
    float cb1[16], cb2[4];
#pragma unroll
    for (int nt = 0; nt < 16; ++nt) cb1[nt] = b1s[nt * 16 + ln];
#pragma unroll
    for (int nt = 0; nt < 4; ++nt)  cb2[nt] = b2s[nt * 16 + ln];

    const float dt  = t[0] / (float)NSTEPS;
    const float dt6 = dt * (1.0f / 6.0f);
    const float dt3 = dt * (1.0f / 3.0f);
    const float hdt = 0.5f * dt;

    // Per-wave 16-row batch tile, held in registers in C/D fragment layout:
    //   element (m, n) with m = r + 8*hi, n = nt*16 + ln  ->  reg [nt][r]
    const int rowbase = blockIdx.x * 64 + wave * 16;
    float yv[4][8];
#pragma unroll
    for (int nt = 0; nt < 4; ++nt)
#pragma unroll
        for (int r = 0; r < 8; ++r)
            yv[nt][r] = x[(size_t)(rowbase + r + 8 * hi) * D_IN + nt * 16 + ln];

    // One f-eval: reads stage input from ts, returns k tile in C/D layout.
    auto feval = [&](f32x8 (&kf)[4]) {
        // ---- GEMM1: h[16x256] = t[16x64] @ W1^T + b1, ELU, staged to hs ----
        // A fragment (16x32 bf16): lane = row(ln); elems 0-15 -> K = hi*16+e.
        // One 32B LDS load each.
        bf16x16 a1[2];
#pragma unroll
        for (int kc = 0; kc < 2; ++kc) {
            const __bf16* p = ts + ln * D_IN + kc * 32 + hi * 16;
            a1[kc] = *(const bf16x16*)p;
        }
#pragma unroll
        for (int nt = 0; nt < 16; ++nt) {
            const float cb = cb1[nt];
            f32x8 c = {cb, cb, cb, cb, cb, cb, cb, cb};   // bias rides in C
#pragma unroll
            for (int kc = 0; kc < 2; ++kc)
                c = __builtin_amdgcn_wmma_f32_16x16x32_bf16(
                        false, a1[kc], false, w1f[nt][kc], (short)0, c, false, false);
            // ELU and transpose-stage into hs (row m = r+8*hi, col = nt*16+ln)
#pragma unroll
            for (int r = 0; r < 8; ++r) {
                float v = c[r];
                v = v > 0.0f ? v : (__expf(v) - 1.0f);
                hs[(r + 8 * hi) * H_DIM + nt * 16 + ln] = (__bf16)v;
            }
        }

        // ---- GEMM2: k[16x64] = h[16x256] @ W2^T + b2 ----
#pragma unroll
        for (int nt = 0; nt < 4; ++nt) {
            const float cb = cb2[nt];
            kf[nt] = (f32x8){cb, cb, cb, cb, cb, cb, cb, cb};
        }
#pragma unroll
        for (int kc = 0; kc < 8; ++kc) {
            const __bf16* p = hs + ln * H_DIM + kc * 32 + hi * 16;
            bf16x16 a2 = *(const bf16x16*)p;
#pragma unroll
            for (int nt = 0; nt < 4; ++nt)
                kf[nt] = __builtin_amdgcn_wmma_f32_16x16x32_bf16(
                             false, a2, false, w2f[kc][nt], (short)0, kf[nt], false, false);
        }
    };

    for (int step = 0; step < NSTEPS; ++step) {
        float av[4][8];
        // Stage-1 input = y; also start the RK4 accumulator.
#pragma unroll
        for (int nt = 0; nt < 4; ++nt)
#pragma unroll
            for (int r = 0; r < 8; ++r) {
                av[nt][r] = yv[nt][r];
                ts[(r + 8 * hi) * D_IN + nt * 16 + ln] = (__bf16)yv[nt][r];
            }

        auto accum_and_stage = [&](f32x8 (&kf)[4], float w, float a, bool do_stage) {
#pragma unroll
            for (int nt = 0; nt < 4; ++nt)
#pragma unroll
                for (int r = 0; r < 8; ++r) {
                    const float k = kf[nt][r];
                    av[nt][r] += w * k;
                    if (do_stage)
                        ts[(r + 8 * hi) * D_IN + nt * 16 + ln] =
                            (__bf16)(yv[nt][r] + a * k);
                }
        };

        f32x8 kf[4];
        feval(kf); accum_and_stage(kf, dt6, hdt, true);   // k1
        feval(kf); accum_and_stage(kf, dt3, hdt, true);   // k2
        feval(kf); accum_and_stage(kf, dt3, dt,  true);   // k3
        feval(kf); accum_and_stage(kf, dt6, 0.0f, false); // k4

#pragma unroll
        for (int nt = 0; nt < 4; ++nt)
#pragma unroll
            for (int r = 0; r < 8; ++r)
                yv[nt][r] = av[nt][r];
    }

    // Final state -> out [B][64] f32 (coalesced per (nt, r) across lanes).
#pragma unroll
    for (int nt = 0; nt < 4; ++nt)
#pragma unroll
        for (int r = 0; r < 8; ++r)
            out[(size_t)(rowbase + r + 8 * hi) * D_IN + nt * 16 + ln] = yv[nt][r];
}

extern "C" void kernel_launch(void* const* d_in, const int* in_sizes, int n_in,
                              void* d_out, int out_size, void* d_ws, size_t ws_size,
                              hipStream_t stream) {
    (void)in_sizes; (void)n_in; (void)out_size; (void)d_ws; (void)ws_size;
    const float* x  = (const float*)d_in[0];
    const float* t  = (const float*)d_in[1];
    const float* W1 = (const float*)d_in[2];
    const float* b1 = (const float*)d_in[3];
    const float* W2 = (const float*)d_in[4];
    const float* b2 = (const float*)d_in[5];
    float* out = (float*)d_out;

    // Opt in to >64KB dynamic LDS (gfx1250 WGP supports up to 320KB).
    (void)hipFuncSetAttribute((const void*)node_rk4_kernel,
                              hipFuncAttributeMaxDynamicSharedMemorySize, SMEM_BYTES);

    node_rk4_kernel<<<BATCH / 64, 128, SMEM_BYTES, stream>>>(x, t, W1, b1, W2, b2, out);
}
